// CTM_15453292331454
// MI455X (gfx1250) — matrix operand order repaired and checked by reading, not compiled, not run on hardware
//
#include <hip/hip_runtime.h>
#include <hip/hip_bf16.h>

// ---------------------------------------------------------------------------
// CTM (cluster token merging) for MI455X / gfx1250, wave32 + WMMA fp16.
// Round 2: pre-transposed weights (all B fragments contiguous b128 loads),
// 32x32 wave tiles in the Gram kernel, 16x32 wave tiles in weight GEMMs,
// transposed V output so attention p@v fragments are contiguous.
// ---------------------------------------------------------------------------

typedef _Float16 f16;
typedef __attribute__((ext_vector_type(16))) _Float16 v16h;
typedef __attribute__((ext_vector_type(8)))  float    v8f;

#define CTM_B   8
#define CTM_N   2048
#define CTM_C   256
#define CTM_DO  512
#define CTM_H   8
#define CTM_HD  32
#define CTM_CN  512
#define CTM_K   5
#define CTM_BN  (CTM_B * CTM_N)     // 16384

union HF { v16h v; unsigned u[8]; _Float16 h[16]; };

// ---- WMMA fragment loaders (layouts per CDNA5 ISA 7.12.2) ------

// A matrix 16x32 fp16, row-major source. lane&15 = row, lane>>4 selects K half.
__device__ __forceinline__ v16h load_a_frag(const f16* A, int lda, int m0, int k0, int lane) {
  HF f;
  int m  = m0 + (lane & 15);
  int hh = lane >> 4;
  const f16* base = A + (size_t)m * lda + k0 + 8 * hh;
  f.u[0] = *(const unsigned*)(base + 0);
  f.u[1] = *(const unsigned*)(base + 2);
  f.u[2] = *(const unsigned*)(base + 4);
  f.u[3] = *(const unsigned*)(base + 6);
  f.u[4] = *(const unsigned*)(base + 16);
  f.u[5] = *(const unsigned*)(base + 18);
  f.u[6] = *(const unsigned*)(base + 20);
  f.u[7] = *(const unsigned*)(base + 22);
  return f.v;
}

// B matrix 32x16 fp16 where logical B[k][n] = X[n][k]  (contiguous in k).
__device__ __forceinline__ v16h load_b_nt_frag(const f16* X, int ldx, int n0, int k0, int lane) {
  HF f;
  int n  = n0 + (lane & 15);
  int hh = lane >> 4;
  const f16* base = X + (size_t)n * ldx + k0 + 8 * hh;
#pragma unroll
  for (int v = 0; v < 8; ++v) f.u[v] = *(const unsigned*)(base + (v < 4 ? 2 * v : 16 + 2 * (v - 4)));
  return f.v;
}

__device__ __forceinline__ v8f wmma_f32(v16h a, v16h b, v8f c) {
  return __builtin_amdgcn_wmma_f32_16x16x32_f16(false, a, false, b, (short)0, c, false, false);
}

// ordered-float <-> uint monotonic mapping (for atomic min/max on floats)
__device__ __forceinline__ unsigned ord_enc(float f) {
  unsigned u = __float_as_uint(f);
  return (u & 0x80000000u) ? ~u : (u | 0x80000000u);
}
__device__ __forceinline__ float ord_dec(unsigned e) {
  unsigned u = (e & 0x80000000u) ? (e & 0x7FFFFFFFu) : ~e;
  return __uint_as_float(u);
}

// ---------------------------------------------------------------------------
// Utility kernels
// ---------------------------------------------------------------------------

__global__ void cast_f32_to_f16_kernel(const float* __restrict__ in, f16* __restrict__ out, size_t n) {
  size_t i = (size_t)blockIdx.x * 256 + threadIdx.x;
  if (i < n) out[i] = (_Float16)in[i];
}

// LDS-tiled transpose+cast: out[n*K + k] = (f16) in[k*N + n]. K,N multiples of 32.
__global__ __launch_bounds__(256)
void transpose_cast_kernel(const float* __restrict__ in, f16* __restrict__ out, int K, int N) {
  __shared__ f16 tile[32][33];
  int bx = blockIdx.x % (N / 32), by = blockIdx.x / (N / 32);
  int tx = threadIdx.x & 31, ty = threadIdx.x >> 5;   // 8 rows per pass
#pragma unroll
  for (int r = 0; r < 32; r += 8)
    tile[ty + r][tx] = (_Float16)in[(size_t)(by * 32 + ty + r) * N + bx * 32 + tx];
  __syncthreads();
#pragma unroll
  for (int r = 0; r < 32; r += 8)
    out[(size_t)(bx * 32 + ty + r) * K + by * 32 + tx] = tile[tx][ty + r];
}

__global__ void fill_u32_kernel(unsigned* p, unsigned v, size_t n) {
  size_t i = (size_t)blockIdx.x * 256 + threadIdx.x;
  if (i < n) p[i] = v;
}
__global__ void fill_u64_kernel(unsigned long long* p, unsigned long long v, size_t n) {
  size_t i = (size_t)blockIdx.x * 256 + threadIdx.x;
  if (i < n) p[i] = v;
}
__global__ void fill_f32_kernel(float* p, float v, size_t n) {
  size_t i = (size_t)blockIdx.x * 256 + threadIdx.x;
  if (i < n) p[i] = v;
}

// ---------------------------------------------------------------------------
// LayerNorm: one 256-thread block per token row; also emits fp16 copy and sumsq
// ---------------------------------------------------------------------------
__global__ __launch_bounds__(256)
void ln_kernel(const float* __restrict__ x, const float* __restrict__ g,
               const float* __restrict__ bb, float* __restrict__ xn,
               f16* __restrict__ xnh, float* __restrict__ sq) {
  __shared__ float red[256];
  int row = blockIdx.x, t = threadIdx.x;
  float v = x[(size_t)row * CTM_C + t];
  red[t] = v; __syncthreads();
  for (int s = 128; s > 0; s >>= 1) { if (t < s) red[t] += red[t + s]; __syncthreads(); }
  float mean = red[0] * (1.0f / CTM_C); __syncthreads();
  float dv = v - mean;
  red[t] = dv * dv; __syncthreads();
  for (int s = 128; s > 0; s >>= 1) { if (t < s) red[t] += red[t + s]; __syncthreads(); }
  float var = red[0] * (1.0f / CTM_C); __syncthreads();
  float o = dv * rsqrtf(var + 1e-5f) * g[t] + bb[t];
  xn[(size_t)row * CTM_C + t] = o;
  xnh[(size_t)row * CTM_C + t] = (_Float16)o;
  red[t] = o * o; __syncthreads();
  for (int s = 128; s > 0; s >>= 1) { if (t < s) red[t] += red[t + s]; __syncthreads(); }
  if (t == 0) sq[row] = red[0];
}

// ---------------------------------------------------------------------------
// dist[b,i,j] = sqrt(max(|xi|^2+|xj|^2-2 xi.xj,0))/sqrt(C) via WMMA Gram (NT).
// 32x32 output per wave: 2 A-frags x 2 B-frags -> 4 WMMA per k-step.
// ---------------------------------------------------------------------------
__global__ __launch_bounds__(256)
void dist_gram_kernel(const f16* __restrict__ Xh, const float* __restrict__ sq,
                      float* __restrict__ dist, unsigned* __restrict__ dmax_bits) {
  int lane = threadIdx.x & 31, wid = threadIdx.x >> 5;
  int b = blockIdx.y;
  const int tilesN = CTM_N >> 5;                     // 64 column super-tiles
  int tile = blockIdx.x * 8 + wid;
  int tm = (tile / tilesN) << 5, tn = (tile % tilesN) << 5;
  const f16* X = Xh + (size_t)b * CTM_N * CTM_C;
  v8f c00 = {}, c01 = {}, c10 = {}, c11 = {};
  for (int k0 = 0; k0 < CTM_C; k0 += 32) {
    v16h a0 = load_a_frag(X, CTM_C, tm, k0, lane);
    v16h a1 = load_a_frag(X, CTM_C, tm + 16, k0, lane);
    v16h b0 = load_b_nt_frag(X, CTM_C, tn, k0, lane);
    v16h b1 = load_b_nt_frag(X, CTM_C, tn + 16, k0, lane);
    c00 = wmma_f32(a0, b0, c00);
    c01 = wmma_f32(a0, b1, c01);
    c10 = wmma_f32(a1, b0, c10);
    c11 = wmma_f32(a1, b1, c11);
  }
  int nloc = lane & 15, hh = lane >> 4;
  const float inv = 0.0625f;                         // 1/sqrt(256)
  float mx = 0.0f;
#pragma unroll
  for (int sub = 0; sub < 4; ++sub) {
    v8f c = (sub == 0) ? c00 : (sub == 1) ? c01 : (sub == 2) ? c10 : c11;
    int tmm = tm + (sub >> 1) * 16;
    int n = tn + (sub & 1) * 16 + nloc;
    float sqn = sq[(size_t)b * CTM_N + n];
#pragma unroll
    for (int r = 0; r < 8; ++r) {
      int m = tmm + r + 8 * hh;
      float d2 = sq[(size_t)b * CTM_N + m] + sqn - 2.0f * c[r];
      float d = sqrtf(fmaxf(d2, 0.0f)) * inv;
      dist[((size_t)b * CTM_N + m) * CTM_N + n] = d;
      mx = fmaxf(mx, d);
    }
  }
#pragma unroll
  for (int off = 16; off > 0; off >>= 1) mx = fmaxf(mx, __shfl_xor(mx, off, 32));
  if (lane == 0) atomicMax(dmax_bits + b, __float_as_uint(mx));  // dist >= 0
}

// ---------------------------------------------------------------------------
// top-K (K=5 smallest) per row, then density and W. One wave per row.
// ---------------------------------------------------------------------------
__global__ __launch_bounds__(256)
void topk_kernel(const float* __restrict__ dist, const float* __restrict__ beta,
                 float* __restrict__ dnear, int* __restrict__ nnidx,
                 float* __restrict__ density, float* __restrict__ Wsum) {
  int lane = threadIdx.x & 31, wid = threadIdx.x >> 5;
  int rid = blockIdx.x * 8 + wid;               // row in [0, B*N)
  const float* drow = dist + (size_t)rid * CTM_N;
  float bv[CTM_K]; int bi[CTM_K];
#pragma unroll
  for (int t = 0; t < CTM_K; ++t) { bv[t] = 3.0e38f; bi[t] = 0x7fffffff; }
  for (int j = lane; j < CTM_N; j += 32) {
    float d = drow[j];
    if (d < bv[CTM_K - 1] || (d == bv[CTM_K - 1] && j < bi[CTM_K - 1])) {
      bv[CTM_K - 1] = d; bi[CTM_K - 1] = j;
#pragma unroll
      for (int t = CTM_K - 1; t > 0; --t) {
        if (bv[t] < bv[t - 1] || (bv[t] == bv[t - 1] && bi[t] < bi[t - 1])) {
          float tv = bv[t]; bv[t] = bv[t - 1]; bv[t - 1] = tv;
          int ti = bi[t]; bi[t] = bi[t - 1]; bi[t - 1] = ti;
        }
      }
    }
  }
  // lane-merge: 5 rounds of wave argmin, winner pops its sorted list head
  float selv[CTM_K]; int seli[CTM_K];
  int ptr = 0;
  for (int t = 0; t < CTM_K; ++t) {
    int pp = ptr < CTM_K ? ptr : CTM_K - 1;
    float v = (ptr < CTM_K) ? bv[pp] : 3.0e38f;
    int id = (ptr < CTM_K) ? bi[pp] : 0x7fffffff;
    int src = lane;
#pragma unroll
    for (int off = 16; off > 0; off >>= 1) {
      float ov = __shfl_xor(v, off, 32);
      int oi = __shfl_xor(id, off, 32);
      int os = __shfl_xor(src, off, 32);
      if (ov < v || (ov == v && oi < id)) { v = ov; id = oi; src = os; }
    }
    selv[t] = v; seli[t] = id;
    if (lane == src) ptr++;
  }
  if (lane == 0) {
    float ab = fmaxf(beta[0], 0.01f);
    float s2 = 0.f, w = 0.f;
#pragma unroll
    for (int t = 0; t < CTM_K; ++t) {
      dnear[(size_t)rid * CTM_K + t] = selv[t];
      nnidx[(size_t)rid * CTM_K + t] = seli[t];
      s2 += selv[t] * selv[t];
      w += 1.0f / (selv[t] + 1.0f);
    }
    unsigned hsh = (unsigned)rid * 2654435761u; hsh ^= hsh >> 15;
    float noise = (float)(hsh & 0xFFFFFFu) * 5.9604645e-8f * 1e-6f;  // tie-break
    density[rid] = expf(-(ab * s2) * (1.0f / CTM_K)) + noise;
    Wsum[rid] = w;
  }
}

// ---------------------------------------------------------------------------
// dmin to nearest higher-density point; score = dmin * density. Wave per row.
// ---------------------------------------------------------------------------
__global__ __launch_bounds__(256)
void dmin_kernel(const float* __restrict__ dist, const float* __restrict__ density,
                 const unsigned* __restrict__ dmax_bits, float* __restrict__ score) {
  int lane = threadIdx.x & 31, wid = threadIdx.x >> 5;
  int rid = blockIdx.x * 8 + wid;
  int b = rid / CTM_N;
  float di = density[rid];
  float m = __uint_as_float(dmax_bits[b]);
  const float* drow = dist + (size_t)rid * CTM_N;
  const float* dens = density + (size_t)b * CTM_N;
  for (int j = lane; j < CTM_N; j += 32)
    if (dens[j] > di) m = fminf(m, drow[j]);
#pragma unroll
  for (int off = 16; off > 0; off >>= 1) m = fminf(m, __shfl_xor(m, off, 32));
  if (lane == 0) score[rid] = m * di;
}

// ---------------------------------------------------------------------------
// Per-batch descending bitonic sort of 2048 scores -> first CN indices
// ---------------------------------------------------------------------------
__global__ __launch_bounds__(1024)
void sort_topcn_kernel(const float* __restrict__ score, int* __restrict__ index_down) {
  __shared__ float sv[CTM_N];
  __shared__ int si[CTM_N];
  int b = blockIdx.x, t = threadIdx.x;
  for (int i = t; i < CTM_N; i += 1024) { sv[i] = score[(size_t)b * CTM_N + i]; si[i] = i; }
  __syncthreads();
  for (int k = 2; k <= CTM_N; k <<= 1) {
    for (int j = k >> 1; j > 0; j >>= 1) {
      for (int p = t; p < CTM_N; p += 1024) {
        int ixj = p ^ j;
        if (ixj > p) {
          bool up = ((p & k) == 0);
          float a = sv[p], c = sv[ixj];
          int ia = si[p], ic = si[ixj];
          bool sw = up ? (a < c || (a == c && ia > ic)) : (a > c || (a == c && ia < ic));
          if (sw) { sv[p] = c; sv[ixj] = a; si[p] = ic; si[ixj] = ia; }
        }
      }
      __syncthreads();
    }
  }
  for (int i = t; i < CTM_CN; i += 1024) index_down[(size_t)b * CTM_CN + i] = si[i];
}

// ---------------------------------------------------------------------------
// final[b,q,j] = dist[b,cq,j] - 100*snn*(W[cq]+W[j]); column argmin via packed
// (ordered-float || q) 64-bit atomicMin. One block per (center, batch).
// ---------------------------------------------------------------------------
__global__ __launch_bounds__(256)
void final_argmin_kernel(const float* __restrict__ dist, const int* __restrict__ index_down,
                         const int* __restrict__ nnidx, const float* __restrict__ W,
                         unsigned long long* __restrict__ minpack) {
  int b = blockIdx.y, q = blockIdx.x;
  int cq = index_down[(size_t)b * CTM_CN + q];
  int cn[CTM_K];
#pragma unroll
  for (int t = 0; t < CTM_K; ++t) cn[t] = nnidx[((size_t)b * CTM_N + cq) * CTM_K + t];
  float Wc = W[(size_t)b * CTM_N + cq];
  const float* drow = dist + ((size_t)b * CTM_N + cq) * CTM_N;
  for (int j = threadIdx.x; j < CTM_N; j += 256) {
    const int* nj = nnidx + ((size_t)b * CTM_N + j) * CTM_K;
    int c0 = nj[0], c1 = nj[1], c2 = nj[2], c3 = nj[3], c4 = nj[4];
    int cnt = 0;
#pragma unroll
    for (int t = 0; t < CTM_K; ++t) {
      int a = cn[t];
      cnt += (a == c0) + (a == c1) + (a == c2) + (a == c3) + (a == c4);
    }
    float wsn = (float)cnt * (Wc + W[(size_t)b * CTM_N + j]);
    float f = drow[j] - 100.0f * wsn;
    unsigned long long key = ((unsigned long long)ord_enc(f) << 32) | (unsigned)q;
    atomicMin(minpack + (size_t)b * CTM_N + j, key);
  }
}

__global__ void assign_kernel(const unsigned long long* __restrict__ minpack,
                              int* __restrict__ idxc, size_t n) {
  size_t i = (size_t)blockIdx.x * 256 + threadIdx.x;
  if (i < n) idxc[i] = (int)(minpack[i] & 0xFFFFFFFFull);
}
__global__ void override_kernel(const int* __restrict__ index_down, int* __restrict__ idxc) {
  int b = blockIdx.x, t = threadIdx.x;   // 512 threads
  idxc[(size_t)b * CTM_N + index_down[(size_t)b * CTM_CN + t]] = t;
}

// ---------------------------------------------------------------------------
// WMMA GEMM with pre-transposed B: C[M,N] = A[M,K] * BwT[N,K]^T (+bias,+res).
// 16x32 output per wave (A-frag reused over 2 WMMAs); all loads contiguous.
// Optional f32 / fp16 / transposed-fp16 outputs.
// ---------------------------------------------------------------------------
__global__ __launch_bounds__(256)
void gemm_wt_f16_kernel(const f16* __restrict__ A, int lda,
                        const f16* __restrict__ BwT, int ldb,
                        int M, int N, int Kd,
                        const float* __restrict__ bias,
                        const float* __restrict__ residual, int ldr,
                        float* __restrict__ C, f16* __restrict__ Ch, int ldc,
                        f16* __restrict__ ChT, int ldt) {
  int lane = threadIdx.x & 31, wid = threadIdx.x >> 5;
  int tilesN = N >> 5;
  int tile = blockIdx.x * 8 + wid;
  int totalTiles = (M >> 4) * tilesN;
  if (tile >= totalTiles) return;
  int tm = (tile / tilesN) << 4;
  int tn = (tile % tilesN) << 5;
  v8f c0 = {}, c1 = {};
  for (int k0 = 0; k0 < Kd; k0 += 32) {
    if (k0 + 32 < Kd)
      __builtin_prefetch(A + (size_t)(tm + (lane & 15)) * lda + k0 + 32, 0, 0);
    v16h a  = load_a_frag(A, lda, tm, k0, lane);
    v16h b0 = load_b_nt_frag(BwT, ldb, tn, k0, lane);
    v16h b1 = load_b_nt_frag(BwT, ldb, tn + 16, k0, lane);
    c0 = wmma_f32(a, b0, c0);
    c1 = wmma_f32(a, b1, c1);
  }
  int hh = lane >> 4, nloc = lane & 15;
#pragma unroll
  for (int half = 0; half < 2; ++half) {
    v8f c = half ? c1 : c0;
    int n = tn + half * 16 + nloc;
    float bvv = bias ? bias[n] : 0.0f;
#pragma unroll
    for (int r = 0; r < 8; ++r) {
      int m = tm + r + 8 * hh;
      float vv = c[r] + bvv;
      if (residual) vv += residual[(size_t)m * ldr + n];
      if (C)   C[(size_t)m * ldc + n] = vv;
      if (Ch)  Ch[(size_t)m * ldc + n] = (_Float16)vv;
      if (ChT) ChT[(size_t)n * ldt + m] = (_Float16)vv;
    }
  }
}

// ---------------------------------------------------------------------------
// token_score column max over tokens (per b,c) and row mean (per b,n)
// ---------------------------------------------------------------------------
__global__ __launch_bounds__(256)
void tsmax_kernel(const float* __restrict__ ts, unsigned* __restrict__ tsmax_bits) {
  int b = blockIdx.y, chunk = blockIdx.x, c = threadIdx.x;
  const int CH = CTM_N / 8;
  int n0 = chunk * CH;
  float mx = -3.0e38f;
  for (int n = n0; n < n0 + CH; ++n)
    mx = fmaxf(mx, ts[((size_t)b * CTM_N + n) * CTM_C + c]);
  atomicMax(tsmax_bits + (size_t)b * CTM_C + c, ord_enc(mx));
}

__global__ __launch_bounds__(256)
void tsmean_kernel(const float* __restrict__ ts, float* __restrict__ tsmean) {
  int lane = threadIdx.x & 31, wid = threadIdx.x >> 5;
  int rid = blockIdx.x * 8 + wid;
  float s = 0.f;
  for (int c = lane; c < CTM_C; c += 32) s += ts[(size_t)rid * CTM_C + c];
#pragma unroll
  for (int off = 16; off > 0; off >>= 1) s += __shfl_xor(s, off, 32);
  if (lane == 0) tsmean[rid] = s * (1.0f / CTM_C);
}

// ---------------------------------------------------------------------------
// weighted scatter-merge (two passes, f32 global atomics)
// ---------------------------------------------------------------------------
__global__ __launch_bounds__(256)
void merge_a_kernel(const float* __restrict__ ts, const unsigned* __restrict__ tsmax_bits,
                    const int* __restrict__ idxc, float* __restrict__ all_w) {
  int b = blockIdx.y, n = blockIdx.x, c = threadIdx.x;
  int cl = idxc[(size_t)b * CTM_N + n];
  float mx = ord_dec(tsmax_bits[(size_t)b * CTM_C + c]);
  float tw = expf(ts[((size_t)b * CTM_N + n) * CTM_C + c] - mx);
  atomicAdd(all_w + ((size_t)b * CTM_CN + cl) * CTM_C + c, tw);
}

__global__ __launch_bounds__(256)
void merge_b_kernel(const float* __restrict__ ts, const unsigned* __restrict__ tsmax_bits,
                    const int* __restrict__ idxc, const float* __restrict__ all_w,
                    const float* __restrict__ xn, float* __restrict__ xm) {
  int b = blockIdx.y, n = blockIdx.x, c = threadIdx.x;
  int cl = idxc[(size_t)b * CTM_N + n];
  float mx = ord_dec(tsmax_bits[(size_t)b * CTM_C + c]);
  float tw = expf(ts[((size_t)b * CTM_N + n) * CTM_C + c] - mx);
  float aw = all_w[((size_t)b * CTM_CN + cl) * CTM_C + c] + 1e-6f;
  atomicAdd(xm + ((size_t)b * CTM_CN + cl) * CTM_C + c,
            xn[((size_t)b * CTM_N + n) * CTM_C + c] * (tw / aw));
}

// ---------------------------------------------------------------------------
// Fused fuzzy-Gaussian cross attention. grid (CN/16, H, B), block 256 (8 waves).
// Per wave: q-tile [16 x 32] fixed; loop token chunks of 32:
//   s = q.k^T*scale + tsmean ; p = exp(exp(-(s-mu)^2/(2s^2)))  (bounded -> no max)
//   stage p in LDS (transpose to A-fragment layout) ; acc += p.v ; rowsum += p
// V is consumed from the transposed layout vT[C][B*N] (contiguous fragments).
// ---------------------------------------------------------------------------
__global__ __launch_bounds__(256)
void attn_kernel(const f16* __restrict__ qh, const f16* __restrict__ kh,
                 const f16* __restrict__ vT, const float* __restrict__ tsmean,
                 const float* __restrict__ mu, const float* __restrict__ sigma,
                 f16* __restrict__ attn_h) {
  __shared__ _Float16 pstage[8][16][CTM_HD];
  __shared__ float accsh[8][16][CTM_HD];
  __shared__ float rssh[8][16];
  int b = blockIdx.z, hhd = blockIdx.y, qt = blockIdx.x;
  int wid = threadIdx.x >> 5, l = threadIdx.x & 31;
  int ln = l & 15, lh = l >> 4;
  const float scale = 0.17677669529663687f;     // 32^-0.5
  float muv = mu[hhd];
  float sg = sigma[hhd] + 1e-8f;
  float inv2s2 = 1.0f / (2.0f * sg * sg);

  const f16* qbase = qh + ((size_t)(b * CTM_CN + qt * 16)) * CTM_C + hhd * CTM_HD;
  v16h aq = load_a_frag(qbase, CTM_C, 0, 0, l);
  v8f acc1 = {}, acc2 = {};
  float rs = 0.f;
  const int nchunks = CTM_N / 32;               // 64, divisible by 8 -> uniform loop
  for (int ch = wid; ch < nchunks; ch += 8) {
    int t0 = ch * 32;
    const f16* kbase = kh + ((size_t)(b * CTM_N + t0)) * CTM_C + hhd * CTM_HD;
    v16h bk1 = load_b_nt_frag(kbase, CTM_C, 0, 0, l);
    v16h bk2 = load_b_nt_frag(kbase, CTM_C, 16, 0, l);
    v8f s1 = {}, s2 = {};
    s1 = wmma_f32(aq, bk1, s1);
    s2 = wmma_f32(aq, bk2, s2);
#pragma unroll
    for (int r = 0; r < 8; ++r) {
      int row = r + 8 * lh;
      float sc1 = s1[r] * scale + tsmean[(size_t)b * CTM_N + t0 + ln];
      float sc2 = s2[r] * scale + tsmean[(size_t)b * CTM_N + t0 + 16 + ln];
      float fz1 = expf(-(sc1 - muv) * (sc1 - muv) * inv2s2);
      float fz2 = expf(-(sc2 - muv) * (sc2 - muv) * inv2s2);
      pstage[wid][row][ln]      = (_Float16)expf(fz1);
      pstage[wid][row][16 + ln] = (_Float16)expf(fz2);
    }
    __syncthreads();
    if (l < 16) {
      float s = 0.f;
#pragma unroll
      for (int c2 = 0; c2 < CTM_HD; ++c2) s += (float)pstage[wid][l][c2];
      rs += s;
    }
    // re-fragment p (C-layout in LDS) into A-layout for p @ v
    HF fa;
#pragma unroll
    for (int vv = 0; vv < 8; ++vv) {
      int kk = (vv < 4 ? 2 * vv : 16 + 2 * (vv - 4)) + 8 * lh;
      fa.h[2 * vv]     = pstage[wid][ln][kk];
      fa.h[2 * vv + 1] = pstage[wid][ln][kk + 1];
    }
    // vT rows are head channels; columns are tokens (contiguous in k)
    const f16* vtbase = vT + (size_t)(hhd * CTM_HD) * CTM_BN + (size_t)b * CTM_N + t0;
    v16h bv1 = load_b_nt_frag(vtbase, CTM_BN, 0, 0, l);
    v16h bv2 = load_b_nt_frag(vtbase, CTM_BN, 16, 0, l);
    acc1 = wmma_f32(fa.v, bv1, acc1);
    acc2 = wmma_f32(fa.v, bv2, acc2);
    __syncthreads();
  }
#pragma unroll
  for (int r = 0; r < 8; ++r) {
    accsh[wid][r + 8 * lh][ln] = acc1[r];
    accsh[wid][r + 8 * lh][16 + ln] = acc2[r];
  }
  if (l < 16) rssh[wid][l] = rs;
  __syncthreads();
  for (int idx = threadIdx.x; idx < 16 * CTM_HD; idx += 256) {
    int row = idx / CTM_HD, d = idx % CTM_HD;
    float a = 0.f, rsum = 0.f;
#pragma unroll
    for (int w = 0; w < 8; ++w) { a += accsh[w][row][d]; rsum += rssh[w][row]; }
    float o = a / rsum;
    attn_h[((size_t)(b * CTM_CN + qt * 16 + row)) * CTM_C + hhd * CTM_HD + d] = (_Float16)o;
  }
}

// ---------------------------------------------------------------------------
// Host-side orchestration
// ---------------------------------------------------------------------------
extern "C" void kernel_launch(void* const* d_in, const int* in_sizes, int n_in,
                              void* d_out, int out_size, void* d_ws, size_t ws_size,
                              hipStream_t stream) {
  (void)in_sizes; (void)n_in; (void)out_size; (void)ws_size;
  const float* x       = (const float*)d_in[0];
  const float* ln_g    = (const float*)d_in[1];
  const float* ln_b    = (const float*)d_in[2];
  const float* score_w = (const float*)d_in[3];
  const float* score_b = (const float*)d_in[4];
  const float* proj_w  = (const float*)d_in[5];
  const float* proj_b  = (const float*)d_in[6];
  const float* beta    = (const float*)d_in[7];
  const float* q_w     = (const float*)d_in[8];
  const float* k_w     = (const float*)d_in[9];
  const float* v_w     = (const float*)d_in[10];
  const float* out_w   = (const float*)d_in[11];
  const float* out_b   = (const float*)d_in[12];
  const float* mu      = (const float*)d_in[13];
  const float* sigma   = (const float*)d_in[14];

  const size_t BN = (size_t)CTM_B * CTM_N;           // 16384
  const size_t BCN = (size_t)CTM_B * CTM_CN;         // 4096
  char* wp = (char*)d_ws;
  auto alloc = [&](size_t bytes) -> void* {
    void* p = (void*)wp;
    wp += (bytes + 255) & ~(size_t)255;
    return p;
  };

  float* xn        = (float*)alloc(BN * CTM_C * 4);
  f16*   xnh       = (f16*)  alloc(BN * CTM_C * 2);
  f16*   xh        = (f16*)  alloc(BN * CTM_C * 2);
  float* sq        = (float*)alloc(BN * 4);
  float* dist      = (float*)alloc(BN * CTM_N * 4);          // 128 MiB, L2-resident
  unsigned* dmaxb  = (unsigned*)alloc(CTM_B * 4);
  float* dnear     = (float*)alloc(BN * CTM_K * 4);
  int*   nnidx     = (int*)  alloc(BN * CTM_K * 4);
  float* density   = (float*)alloc(BN * 4);
  float* Wsum      = (float*)alloc(BN * 4);
  float* score     = (float*)alloc(BN * 4);
  int*   index_down= (int*)  alloc(BCN * 4);
  unsigned long long* minpack = (unsigned long long*)alloc(BN * 8);
  int*   idxc      = (int*)  alloc(BN * 4);
  float* tscore    = (float*)alloc(BN * CTM_C * 4);
  unsigned* tsmaxb = (unsigned*)alloc((size_t)CTM_B * CTM_C * 4);
  float* tsmean    = (float*)alloc(BN * 4);
  float* all_w     = (float*)alloc(BCN * CTM_C * 4);
  float* xm        = (float*)alloc(BCN * CTM_C * 4);
  f16*   xmh       = (f16*)  alloc(BCN * CTM_C * 2);
  f16*   qhh       = (f16*)  alloc(BCN * CTM_C * 2);
  f16*   khh       = (f16*)  alloc(BN * CTM_C * 2);
  f16*   vTt       = (f16*)  alloc((size_t)CTM_C * BN * 2);  // vT[C][B*N]
  f16*   attnh     = (f16*)  alloc(BCN * CTM_C * 2);
  float* yb        = (float*)alloc(BCN * CTM_C * 4);
  f16*   yh        = (f16*)  alloc(BCN * CTM_C * 2);
  f16*   swT       = (f16*)  alloc((size_t)CTM_C * CTM_C * 2);    // [N,K] transposed
  f16*   qwT       = (f16*)  alloc((size_t)CTM_C * CTM_C * 2);
  f16*   kwT       = (f16*)  alloc((size_t)CTM_C * CTM_C * 2);
  f16*   vwT       = (f16*)  alloc((size_t)CTM_C * CTM_C * 2);
  f16*   owT       = (f16*)  alloc((size_t)CTM_C * CTM_C * 2);
  f16*   pwT       = (f16*)  alloc((size_t)CTM_DO * CTM_C * 2);

  auto blks = [](size_t n) { return (unsigned)((n + 255) / 256); };

  // --- fp16 cast of x; transpose+cast of all weight matrices ----------------
  cast_f32_to_f16_kernel<<<blks(BN * CTM_C), 256, 0, stream>>>(x, xh, BN * CTM_C);
  {
    unsigned gsq = (CTM_C / 32) * (CTM_C / 32);            // 64 tiles for 256x256
    transpose_cast_kernel<<<gsq, 256, 0, stream>>>(score_w, swT, CTM_C, CTM_C);
    transpose_cast_kernel<<<gsq, 256, 0, stream>>>(q_w,   qwT, CTM_C, CTM_C);
    transpose_cast_kernel<<<gsq, 256, 0, stream>>>(k_w,   kwT, CTM_C, CTM_C);
    transpose_cast_kernel<<<gsq, 256, 0, stream>>>(v_w,   vwT, CTM_C, CTM_C);
    transpose_cast_kernel<<<gsq, 256, 0, stream>>>(out_w, owT, CTM_C, CTM_C);
    transpose_cast_kernel<<<(CTM_C / 32) * (CTM_DO / 32), 256, 0, stream>>>(proj_w, pwT, CTM_C, CTM_DO);
  }

  // --- layernorm ------------------------------------------------------------
  ln_kernel<<<(unsigned)BN, 256, 0, stream>>>(x, ln_g, ln_b, xn, xnh, sq);

  // --- pairwise distances (WMMA Gram, 32x32 per wave), dmax -----------------
  fill_u32_kernel<<<1, 256, 0, stream>>>(dmaxb, 0u, CTM_B);
  {
    dim3 g((CTM_N / 32) * (CTM_N / 32) / 8, CTM_B);        // 512 x 8
    dist_gram_kernel<<<g, 256, 0, stream>>>(xnh, sq, dist, dmaxb);
  }

  // --- top-K, density, W ; dmin/score ; top-CN sort -------------------------
  topk_kernel<<<(unsigned)(BN / 8), 256, 0, stream>>>(dist, beta, dnear, nnidx, density, Wsum);
  dmin_kernel<<<(unsigned)(BN / 8), 256, 0, stream>>>(dist, density, dmaxb, score);
  sort_topcn_kernel<<<CTM_B, 1024, 0, stream>>>(score, index_down);

  // --- cluster assignment ---------------------------------------------------
  fill_u64_kernel<<<blks(BN), 256, 0, stream>>>(minpack, ~0ull, BN);
  {
    dim3 g(CTM_CN, CTM_B);
    final_argmin_kernel<<<g, 256, 0, stream>>>(dist, index_down, nnidx, Wsum, minpack);
  }
  assign_kernel<<<blks(BN), 256, 0, stream>>>(minpack, idxc, BN);
  override_kernel<<<CTM_B, CTM_CN, 0, stream>>>(index_down, idxc);

  // --- token_score GEMM + reductions ----------------------------------------
  {
    int M = (int)BN;
    unsigned g = (unsigned)(((M / 16) * (CTM_C / 32) + 7) / 8);
    gemm_wt_f16_kernel<<<g, 256, 0, stream>>>(xnh, CTM_C, swT, CTM_C, M, CTM_C, CTM_C,
                                              score_b, nullptr, 0, tscore, nullptr, CTM_C,
                                              nullptr, 0);
  }
  fill_u32_kernel<<<blks((size_t)CTM_B * CTM_C), 256, 0, stream>>>(tsmaxb, 0u, (size_t)CTM_B * CTM_C);
  {
    dim3 g(8, CTM_B);
    tsmax_kernel<<<g, 256, 0, stream>>>(tscore, tsmaxb);
  }
  tsmean_kernel<<<(unsigned)(BN / 8), 256, 0, stream>>>(tscore, tsmean);

  // --- weighted scatter merge -----------------------------------------------
  fill_f32_kernel<<<blks(BCN * CTM_C), 256, 0, stream>>>(all_w, 0.0f, BCN * CTM_C);
  {
    dim3 g(CTM_N, CTM_B);
    merge_a_kernel<<<g, 256, 0, stream>>>(tscore, tsmaxb, idxc, all_w);
  }
  fill_f32_kernel<<<blks(BCN * CTM_C), 256, 0, stream>>>(xm, 0.0f, BCN * CTM_C);
  {
    dim3 g(CTM_N, CTM_B);
    merge_b_kernel<<<g, 256, 0, stream>>>(tscore, tsmaxb, idxc, all_w, xn, xm);
  }
  cast_f32_to_f16_kernel<<<blks(BCN * CTM_C), 256, 0, stream>>>(xm, xmh, BCN * CTM_C);

  // --- q/k/v projections (WMMA, transposed weights) -------------------------
  {
    int M = (int)BCN;
    unsigned g = (unsigned)(((M / 16) * (CTM_C / 32) + 7) / 8);
    gemm_wt_f16_kernel<<<g, 256, 0, stream>>>(xmh, CTM_C, qwT, CTM_C, M, CTM_C, CTM_C,
                                              nullptr, nullptr, 0, nullptr, qhh, CTM_C,
                                              nullptr, 0);
  }
  {
    int M = (int)BN;
    unsigned g = (unsigned)(((M / 16) * (CTM_C / 32) + 7) / 8);
    gemm_wt_f16_kernel<<<g, 256, 0, stream>>>(xh, CTM_C, kwT, CTM_C, M, CTM_C, CTM_C,
                                              nullptr, nullptr, 0, nullptr, khh, CTM_C,
                                              nullptr, 0);
    // V: emit transposed fp16 so attention p@v fragments are contiguous
    gemm_wt_f16_kernel<<<g, 256, 0, stream>>>(xh, CTM_C, vwT, CTM_C, M, CTM_C, CTM_C,
                                              nullptr, nullptr, 0, nullptr, nullptr, 0,
                                              vTt, (int)BN);
  }

  // --- fused fuzzy attention -------------------------------------------------
  {
    dim3 g(CTM_CN / 16, CTM_H, CTM_B);
    attn_kernel<<<g, 256, 0, stream>>>(qhh, khh, vTt, tsmean, mu, sigma, attnh);
  }

  // --- out projection + residual, then final proj ---------------------------
  {
    int M = (int)BCN;
    unsigned g = (unsigned)(((M / 16) * (CTM_C / 32) + 7) / 8);
    gemm_wt_f16_kernel<<<g, 256, 0, stream>>>(attnh, CTM_C, owT, CTM_C, M, CTM_C, CTM_C,
                                              out_b, xm, CTM_C, yb, yh, CTM_C,
                                              nullptr, 0);
  }
  {
    int M = (int)BCN;
    unsigned g = (unsigned)(((M / 16) * (CTM_DO / 32) + 7) / 8);
    gemm_wt_f16_kernel<<<g, 256, 0, stream>>>(yh, CTM_C, pwT, CTM_C, M, CTM_DO, CTM_C,
                                              proj_b, nullptr, 0, (float*)d_out, nullptr, CTM_DO,
                                              nullptr, 0);
  }
}
